// TransformerSelfAttention_2723009266487
// MI455X (gfx1250) — compile-verified, matrix operand order checked by reading, not run on hardware
//
#include <hip/hip_runtime.h>
#include <hip/hip_bf16.h>
#include <math.h>

typedef unsigned short u16;
typedef __bf16 bf16x16 __attribute__((ext_vector_type(16)));
typedef float f32x8 __attribute__((ext_vector_type(8)));
typedef unsigned int uint8v __attribute__((ext_vector_type(8)));
typedef int v4i __attribute__((__vector_size__(16)));

#define D_MODEL 1024
#define SEQ 2048
#define NB 2
#define NH 16
#define HD 64
#define ROWS (NB * SEQ)   // 4096

#if __has_builtin(__builtin_amdgcn_global_load_async_to_lds_b128)
#define USE_ASYNC_LDS 1
#endif

__device__ __forceinline__ u16 f32_to_bf16(float f) {
    unsigned int u = __float_as_uint(f);
    u += 0x7FFFu + ((u >> 16) & 1u);   // round to nearest even
    return (u16)(u >> 16);
}

__device__ __forceinline__ bf16x16 ld_frag(const u16* plo, const u16* phi) {
    uint4 lo = *(const uint4*)plo;
    uint4 hi = *(const uint4*)phi;
    uint8v t;
    t[0] = lo.x; t[1] = lo.y; t[2] = lo.z; t[3] = lo.w;
    t[4] = hi.x; t[5] = hi.y; t[6] = hi.z; t[7] = hi.w;
    return __builtin_bit_cast(bf16x16, t);
}

__device__ __forceinline__ f32x8 wmma_bf16(bf16x16 a, bf16x16 b, f32x8 c) {
    return __builtin_amdgcn_wmma_f32_16x16x32_bf16(false, a, false, b, (short)0, c, false, false);
}

__device__ __forceinline__ f32x8 zero8() {
    f32x8 z;
#pragma unroll
    for (int i = 0; i < 8; ++i) z[i] = 0.0f;
    return z;
}

// Copy 16B global -> LDS. Async (memory->LDS direct, ASYNCcnt) when available.
__device__ __forceinline__ void cp16_g2l(const u16* g, u16* l) {
#ifdef USE_ASYNC_LDS
    __builtin_amdgcn_global_load_async_to_lds_b128(
        (__attribute__((address_space(1))) v4i*)const_cast<u16*>(g),
        (__attribute__((address_space(3))) v4i*)l, 0, 0);
#else
    *(uint4*)l = *(const uint4*)g;
#endif
}

__device__ __forceinline__ void wait_async0() {
#ifdef USE_ASYNC_LDS
#if __has_builtin(__builtin_amdgcn_s_wait_asynccnt)
    __builtin_amdgcn_s_wait_asynccnt(0);
#else
    asm volatile("s_wait_asynccnt 0x0" ::: "memory");
#endif
#endif
}

// ---------------------------------------------------------------------------
// Weight cast + transpose: W[K,N] f32 -> out[N,K] bf16.  32x32 LDS tiles,
// coalesced reads and writes.  Grid: (N/32, K/32), 256 threads.
// ---------------------------------------------------------------------------
__global__ __launch_bounds__(256) void cast_transpose_k(const float* __restrict__ W,
                                                        u16* __restrict__ out,
                                                        int K, int N) {
    __shared__ float t[32][33];
    const int n0 = blockIdx.x * 32;
    const int k0 = blockIdx.y * 32;
    const int c = threadIdx.x & 31;
    const int r = threadIdx.x >> 5;   // 0..7
#pragma unroll
    for (int i = 0; i < 4; ++i) {
        int row = r + i * 8;
        t[row][c] = W[(size_t)(k0 + row) * N + n0 + c];
    }
    __syncthreads();
#pragma unroll
    for (int i = 0; i < 4; ++i) {
        int row = r + i * 8;          // row within N-tile
        out[(size_t)(n0 + row) * K + k0 + c] = f32_to_bf16(t[c][row]);
    }
}

// ---------------------------------------------------------------------------
// LayerNorm: one row (D=1024) per 256-thread block. Writes f32 (optional) + bf16.
// ---------------------------------------------------------------------------
__global__ __launch_bounds__(256) void layernorm_k(const float* __restrict__ x,
                                                   const float* __restrict__ g,
                                                   const float* __restrict__ b,
                                                   float* __restrict__ outF,
                                                   u16* __restrict__ outB) {
    __shared__ float r1[256];
    __shared__ float r2[256];
    const int tid = threadIdx.x;
    const size_t row = blockIdx.x;
    const float* xr = x + row * D_MODEL;
    float v[4];
    const int base = tid * 4;
    float s = 0.f, s2 = 0.f;
#pragma unroll
    for (int i = 0; i < 4; ++i) {
        v[i] = xr[base + i];
        s += v[i];
        s2 += v[i] * v[i];
    }
    r1[tid] = s; r2[tid] = s2;
    __syncthreads();
    for (int off = 128; off > 0; off >>= 1) {
        if (tid < off) { r1[tid] += r1[tid + off]; r2[tid] += r2[tid + off]; }
        __syncthreads();
    }
    const float mu = r1[0] * (1.0f / D_MODEL);
    const float var = r2[0] * (1.0f / D_MODEL) - mu * mu;
    const float rs = rsqrtf(var + 1e-5f);
#pragma unroll
    for (int i = 0; i < 4; ++i) {
        int col = base + i;
        float xn = (v[i] - mu) * rs * g[col] + b[col];
        if (outF) outF[row * D_MODEL + col] = xn;
        outB[row * D_MODEL + col] = f32_to_bf16(xn);
    }
}

// ---------------------------------------------------------------------------
// Tiled bf16 GEMM with WMMA.  C[M,N] = A[M,K] * Bt[N,K]^T  (Bt is B transposed)
// Block tile 128x128, 8 waves (4 M x 2 N), each wave 32x64, K step 32.
// Double-buffered LDS; async memory->LDS copies when available.
// ---------------------------------------------------------------------------
#define MODE_Q    0
#define MODE_K    1
#define MODE_V    2
#define MODE_F32  3   // outF = acc*scale + bias + resid
#define MODE_GELU 4   // outB = bf16(gelu(acc*scale + bias))

__global__ __launch_bounds__(256) void gemm_bf16_k(const u16* __restrict__ A,
                                                   const u16* __restrict__ Bt,
                                                   int M, int N, int K,
                                                   int mode, float scale,
                                                   const float* __restrict__ bias,
                                                   const float* __restrict__ resid,
                                                   float* __restrict__ outF,
                                                   u16* __restrict__ outB) {
    __shared__ u16 lA[2][128 * 40];
    __shared__ u16 lB[2][128 * 40];
    const int tid = threadIdx.x;
    const int lane = tid & 31;
    const int wave = tid >> 5;
    const int waveM = wave & 3;       // 0..3
    const int waveN = wave >> 2;      // 0..1
    const int bM = blockIdx.y * 128;
    const int bN = blockIdx.x * 128;
    const int l16 = lane & 15;
    const int khalf = (lane >> 4) * 8;     // 0 or 8 (K sub-chunk base)
    const int mloc = (lane >> 4) * 8;      // 0 or 8 (C-layout row base)
    // staging assignment: this thread copies chunks c = tid*2, tid*2+1
    const int sr0 = (tid * 2) >> 2, sc0 = (tid * 2) & 3;
    const int sr1 = (tid * 2 + 1) >> 2, sc1 = (tid * 2 + 1) & 3;

    f32x8 acc[2][4];
#pragma unroll
    for (int mt = 0; mt < 2; ++mt)
#pragma unroll
        for (int nt = 0; nt < 4; ++nt) acc[mt][nt] = zero8();

    // stage K-slice kk into buffer buf
    auto stage = [&](int buf, int kk) {
        cp16_g2l(A + (size_t)(bM + sr0) * K + kk + sc0 * 8, &lA[buf][sr0 * 40 + sc0 * 8]);
        cp16_g2l(A + (size_t)(bM + sr1) * K + kk + sc1 * 8, &lA[buf][sr1 * 40 + sc1 * 8]);
        cp16_g2l(Bt + (size_t)(bN + sr0) * K + kk + sc0 * 8, &lB[buf][sr0 * 40 + sc0 * 8]);
        cp16_g2l(Bt + (size_t)(bN + sr1) * K + kk + sc1 * 8, &lB[buf][sr1 * 40 + sc1 * 8]);
    };

    stage(0, 0);
    wait_async0();
    __syncthreads();

    int cur = 0;
    for (int kk = 0; kk < K; kk += 32) {
        if (kk + 32 < K) stage(cur ^ 1, kk + 32);   // prefetch next slice
        bf16x16 af[2], bfr[4];
#pragma unroll
        for (int mt = 0; mt < 2; ++mt) {
            const u16* p = &lA[cur][(waveM * 32 + mt * 16 + l16) * 40 + khalf];
            af[mt] = ld_frag(p, p + 16);
        }
#pragma unroll
        for (int nt = 0; nt < 4; ++nt) {
            const u16* p = &lB[cur][(waveN * 64 + nt * 16 + l16) * 40 + khalf];
            bfr[nt] = ld_frag(p, p + 16);
        }
#pragma unroll
        for (int mt = 0; mt < 2; ++mt)
#pragma unroll
            for (int nt = 0; nt < 4; ++nt)
                acc[mt][nt] = wmma_bf16(af[mt], bfr[nt], acc[mt][nt]);
        wait_async0();
        __syncthreads();
        cur ^= 1;
    }

    // Epilogue. C layout: vgpr j -> row mloc+j, lane%16 -> col.
#pragma unroll
    for (int mt = 0; mt < 2; ++mt) {
#pragma unroll
        for (int nt = 0; nt < 4; ++nt) {
            const int gcol = bN + waveN * 64 + nt * 16 + l16;
#pragma unroll
            for (int j = 0; j < 8; ++j) {
                const int grow = bM + waveM * 32 + mt * 16 + mloc + j;
                float v = acc[mt][nt][j] * scale;
                if (mode == MODE_Q || mode == MODE_K) {
                    int bb = grow >> 11, s = grow & 2047;
                    int h = gcol >> 6, d = gcol & 63;
                    outB[(((size_t)(bb * NH + h) * SEQ) + s) * HD + d] = f32_to_bf16(v);
                } else if (mode == MODE_V) {
                    int bb = grow >> 11, s = grow & 2047;
                    int h = gcol >> 6, d = gcol & 63;
                    outB[(((size_t)(bb * NH + h) * HD) + d) * SEQ + s] = f32_to_bf16(v);
                } else if (mode == MODE_F32) {
                    size_t idx = (size_t)grow * N + gcol;
                    outF[idx] = v + bias[gcol] + resid[idx];
                } else { // MODE_GELU
                    float xv = v + bias[gcol];
                    float gl = 0.5f * xv * (1.0f + erff(xv * 0.70710678118654752f));
                    outB[(size_t)grow * N + gcol] = f32_to_bf16(gl);
                }
            }
        }
    }
}

// ---------------------------------------------------------------------------
// Flash attention, WMMA.  q,k: [b,h,s,d] bf16 (q pre-scaled by 1/8).
// vt: [b,h,d,s] bf16.  out yb: [b,s, h*64+d] bf16.
// Block = 128 threads = 4 waves; wave owns 16 query rows; 64-key tiles.
// ---------------------------------------------------------------------------
__global__ __launch_bounds__(128) void attn_k(const u16* __restrict__ q,
                                              const u16* __restrict__ k,
                                              const u16* __restrict__ vt,
                                              u16* __restrict__ yb) {
    __shared__ u16 lP[4][16 * 72];
    const int lane = threadIdx.x & 31;
    const int wave = threadIdx.x >> 5;
    const int bh = blockIdx.y;                 // b*NH + h
    const int qbase = blockIdx.x * 64 + wave * 16;
    const int l16 = lane & 15;
    const int khalf = (lane >> 4) * 8;
    const int mloc = (lane >> 4) * 8;
    const size_t headOff = (size_t)bh * SEQ * HD;

    bf16x16 qf[2];
    {
        const u16* qp = q + headOff + (size_t)(qbase + l16) * HD;
#pragma unroll
        for (int kt = 0; kt < 2; ++kt) {
            const u16* p = qp + kt * 32 + khalf;
            qf[kt] = ld_frag(p, p + 16);
        }
    }

    f32x8 O[4];
    float rmax[8], rsum[8];
#pragma unroll
    for (int n = 0; n < 4; ++n) O[n] = zero8();
#pragma unroll
    for (int j = 0; j < 8; ++j) { rmax[j] = -INFINITY; rsum[j] = 0.0f; }

    for (int kb = 0; kb < SEQ; kb += 64) {
        // ---- scores: S[16 x 64] = Q * K^T (q pre-scaled) ----
        f32x8 sAcc[4];
#pragma unroll
        for (int n = 0; n < 4; ++n) {
            sAcc[n] = zero8();
#pragma unroll
            for (int kt = 0; kt < 2; ++kt) {
                const u16* kp = k + headOff + (size_t)(kb + n * 16 + l16) * HD + kt * 32 + khalf;
                bf16x16 bf = ld_frag(kp, kp + 16);
                sAcc[n] = wmma_bf16(qf[kt], bf, sAcc[n]);
            }
            const int kj = kb + n * 16 + l16;
#pragma unroll
            for (int j = 0; j < 8; ++j) {
                if (qbase + mloc + j == kj) sAcc[n][j] = -INFINITY;
            }
        }
        // ---- online softmax ----
        float corr[8];
#pragma unroll
        for (int j = 0; j < 8; ++j) {
            float t = sAcc[0][j];
#pragma unroll
            for (int n = 1; n < 4; ++n) t = fmaxf(t, sAcc[n][j]);
            t = fmaxf(t, __shfl_xor(t, 1));
            t = fmaxf(t, __shfl_xor(t, 2));
            t = fmaxf(t, __shfl_xor(t, 4));
            t = fmaxf(t, __shfl_xor(t, 8));
            float nm = fmaxf(rmax[j], t);
            corr[j] = expf(rmax[j] - nm);
            rmax[j] = nm;
            rsum[j] *= corr[j];
        }
#pragma unroll
        for (int n = 0; n < 4; ++n)
#pragma unroll
            for (int j = 0; j < 8; ++j) O[n][j] *= corr[j];

        float ps[8];
#pragma unroll
        for (int j = 0; j < 8; ++j) ps[j] = 0.0f;
#pragma unroll
        for (int n = 0; n < 4; ++n)
#pragma unroll
            for (int j = 0; j < 8; ++j) {
                float p = expf(sAcc[n][j] - rmax[j]);
                sAcc[n][j] = p;
                ps[j] += p;
            }
#pragma unroll
        for (int j = 0; j < 8; ++j) {
            float t = ps[j];
            t += __shfl_xor(t, 1);
            t += __shfl_xor(t, 2);
            t += __shfl_xor(t, 4);
            t += __shfl_xor(t, 8);
            rsum[j] += t;
        }
        // ---- P (C layout) -> LDS -> A-fragment layout ----
        u16* base = &lP[wave][0];
#pragma unroll
        for (int n = 0; n < 4; ++n)
#pragma unroll
            for (int j = 0; j < 8; ++j)
                base[(mloc + j) * 72 + n * 16 + l16] = f32_to_bf16(sAcc[n][j]);
        bf16x16 pf[2];
#pragma unroll
        for (int kt = 0; kt < 2; ++kt) {
            const u16* p = base + l16 * 72 + kt * 32 + khalf;
            pf[kt] = ld_frag(p, p + 16);
        }
        // ---- O += P * V  (V^T layout: [d][s], contiguous over keys) ----
#pragma unroll
        for (int nd = 0; nd < 4; ++nd) {
#pragma unroll
            for (int kt = 0; kt < 2; ++kt) {
                const u16* vp = vt + (size_t)bh * HD * SEQ +
                                (size_t)(nd * 16 + l16) * SEQ + kb + kt * 32 + khalf;
                bf16x16 vf = ld_frag(vp, vp + 16);
                O[nd] = wmma_bf16(pf[kt], vf, O[nd]);
            }
        }
    }

    const int bb = bh >> 4, h = bh & 15;
#pragma unroll
    for (int j = 0; j < 8; ++j) rsum[j] = 1.0f / rsum[j];
#pragma unroll
    for (int nd = 0; nd < 4; ++nd)
#pragma unroll
        for (int j = 0; j < 8; ++j) {
            int s = qbase + mloc + j;
            size_t idx = ((size_t)bb * SEQ + s) * D_MODEL + h * HD + nd * 16 + l16;
            yb[idx] = f32_to_bf16(O[nd][j] * rsum[j]);
        }
}

// ---------------------------------------------------------------------------
extern "C" void kernel_launch(void* const* d_in, const int* in_sizes, int n_in,
                              void* d_out, int out_size, void* d_ws, size_t ws_size,
                              hipStream_t stream) {
    (void)in_sizes; (void)n_in; (void)out_size; (void)ws_size;
    const float* x   = (const float*)d_in[0];
    const float* Wk  = (const float*)d_in[1];
    const float* Wq  = (const float*)d_in[2];
    const float* Wv  = (const float*)d_in[3];
    const float* Wp  = (const float*)d_in[4];
    const float* bp  = (const float*)d_in[5];
    const float* g1  = (const float*)d_in[6];
    const float* b1  = (const float*)d_in[7];
    const float* g2  = (const float*)d_in[8];
    const float* b2  = (const float*)d_in[9];
    const float* W1  = (const float*)d_in[10];
    const float* bm1 = (const float*)d_in[11];
    const float* W2  = (const float*)d_in[12];
    const float* bm2 = (const float*)d_in[13];
    float* out = (float*)d_out;

    char* ws = (char*)d_ws;
    size_t off = 0;
    auto alloc = [&](size_t bytes) -> char* {
        char* p = ws + off;
        off += (bytes + 255) & ~(size_t)255;
        return p;
    };
    float* xnF = (float*)alloc((size_t)ROWS * D_MODEL * 4);   // 16 MB
    u16*   xnB = (u16*)  alloc((size_t)ROWS * D_MODEL * 2);   //  8 MB
    u16*   WqT = (u16*)  alloc((size_t)D_MODEL * D_MODEL * 2);
    u16*   WkT = (u16*)  alloc((size_t)D_MODEL * D_MODEL * 2);
    u16*   WvT = (u16*)  alloc((size_t)D_MODEL * D_MODEL * 2);
    u16*   WpT = (u16*)  alloc((size_t)D_MODEL * D_MODEL * 2);
    u16*   W1T = (u16*)  alloc((size_t)D_MODEL * 4 * D_MODEL * 2);  // 8 MB
    u16*   W2T = (u16*)  alloc((size_t)D_MODEL * 4 * D_MODEL * 2);  // 8 MB
    char*  pool = alloc((size_t)ROWS * D_MODEL * 2 * 4);            // 32 MB shared pool
    u16*   qB  = (u16*)pool;
    u16*   kB  = (u16*)(pool + (size_t)ROWS * HD * NH * 2);
    u16*   vtB = (u16*)(pool + (size_t)ROWS * HD * NH * 4);
    u16*   yB  = (u16*)(pool + (size_t)ROWS * HD * NH * 6);
    u16*   uB  = (u16*)pool;   // aliases q/k/vt/y (dead by MLP time)
    float* hF  = (float*)alloc((size_t)ROWS * D_MODEL * 4);   // 16 MB
    u16*   hnB = (u16*)  alloc((size_t)ROWS * D_MODEL * 2);   //  8 MB

    // 1) Weight casts + transposes to bf16 [N,K] (coalesced 32x32 tiles)
    cast_transpose_k<<<dim3(D_MODEL / 32, D_MODEL / 32), 256, 0, stream>>>(Wq, WqT, D_MODEL, D_MODEL);
    cast_transpose_k<<<dim3(D_MODEL / 32, D_MODEL / 32), 256, 0, stream>>>(Wk, WkT, D_MODEL, D_MODEL);
    cast_transpose_k<<<dim3(D_MODEL / 32, D_MODEL / 32), 256, 0, stream>>>(Wv, WvT, D_MODEL, D_MODEL);
    cast_transpose_k<<<dim3(D_MODEL / 32, D_MODEL / 32), 256, 0, stream>>>(Wp, WpT, D_MODEL, D_MODEL);
    cast_transpose_k<<<dim3(4 * D_MODEL / 32, D_MODEL / 32), 256, 0, stream>>>(W1, W1T, D_MODEL, 4 * D_MODEL);
    cast_transpose_k<<<dim3(D_MODEL / 32, 4 * D_MODEL / 32), 256, 0, stream>>>(W2, W2T, 4 * D_MODEL, D_MODEL);

    // 2) LayerNorm1: x -> xn (f32 + bf16)
    layernorm_k<<<ROWS, 256, 0, stream>>>(x, g1, b1, xnF, xnB);

    // 3) QKV projections (q pre-scaled by 1/sqrt(HD))
    dim3 gQKV(D_MODEL / 128, ROWS / 128);
    gemm_bf16_k<<<gQKV, 256, 0, stream>>>(xnB, WqT, ROWS, D_MODEL, D_MODEL,
                                          MODE_Q, 0.125f, nullptr, nullptr, nullptr, qB);
    gemm_bf16_k<<<gQKV, 256, 0, stream>>>(xnB, WkT, ROWS, D_MODEL, D_MODEL,
                                          MODE_K, 1.0f, nullptr, nullptr, nullptr, kB);
    gemm_bf16_k<<<gQKV, 256, 0, stream>>>(xnB, WvT, ROWS, D_MODEL, D_MODEL,
                                          MODE_V, 1.0f, nullptr, nullptr, nullptr, vtB);

    // 4) Flash attention
    attn_k<<<dim3(SEQ / 64, NB * NH), 128, 0, stream>>>(qB, kB, vtB, yB);

    // 5) Output projection + residual: h = xn + y*Wp + bp
    gemm_bf16_k<<<gQKV, 256, 0, stream>>>(yB, WpT, ROWS, D_MODEL, D_MODEL,
                                          MODE_F32, 1.0f, bp, xnF, hF, nullptr);

    // 6) LayerNorm2: h -> hn (bf16 only)
    layernorm_k<<<ROWS, 256, 0, stream>>>(hF, g2, b2, nullptr, hnB);

    // 7) MLP up + exact GELU: u = gelu(hn*W1 + bm1)
    dim3 gUp(4 * D_MODEL / 128, ROWS / 128);
    gemm_bf16_k<<<gUp, 256, 0, stream>>>(hnB, W1T, ROWS, 4 * D_MODEL, D_MODEL,
                                         MODE_GELU, 1.0f, bm1, nullptr, nullptr, uB);

    // 8) MLP down + residual: out = h + u*W2 + bm2
    gemm_bf16_k<<<gQKV, 256, 0, stream>>>(uB, W2T, ROWS, D_MODEL, 4 * D_MODEL,
                                          MODE_F32, 1.0f, bm2, hF, out, nullptr);
}
